// KAN_original_31267361915245
// MI455X (gfx1250) — matrix operand order repaired and checked by reading, not compiled, not run on hardware
//
#include <hip/hip_runtime.h>
#include <math.h>

typedef __attribute__((ext_vector_type(16))) _Float16 v16h;
typedef __attribute__((ext_vector_type(8)))  _Float16 v8h;
typedef __attribute__((ext_vector_type(8)))  float    v8f;

#define B_SZ   128
#define IN1    3072
#define H1     256
#define OUT2   10
#define NOUT2P 16      // padded N for layer-2 GEMM
#define NB     8       // num + k
#define FEAT   9       // silu + 8 basis
#define K1     (IN1 * FEAT)   // 27648
#define K2     (H1  * FEAT)   // 2304

__device__ __forceinline__ float silu_f(float x) {
  return x / (1.0f + expf(-x));
}

__device__ __forceinline__ float selu_f(float x) {
  const float scale = 1.0507009873554805f;
  const float alpha = 1.6732632423543772f;
  return scale * (x > 0.0f ? x : alpha * (expf(x) - 1.0f));
}

// Cox-de Boor, k=3, uniform extended grid: grid[j] = -2.2 + 0.4*j, j = 0..11
__device__ __forceinline__ void bspline8(float x, float* out) {
  float b[11];
  #pragma unroll
  for (int j = 0; j < 11; ++j) {
    float gj = -2.2f + 0.4f * (float)j;
    b[j] = (x >= gj && x < gj + 0.4f) ? 1.0f : 0.0f;
  }
  #pragma unroll
  for (int d = 1; d <= 3; ++d) {
    float inv = 1.0f / (0.4f * (float)d);
    for (int j = 0; j < 11 - d; ++j) {
      float gj    = -2.2f + 0.4f * (float)j;
      float left  = (x - gj) * inv;
      float right = ((gj + 0.4f * (float)(d + 1)) - x) * inv;
      b[j] = left * b[j] + right * b[j + 1];
    }
  }
  #pragma unroll
  for (int g = 0; g < 8; ++g) out[g] = b[g];
}

// ---- layer-1 features: F[(b*IN1+i)*9 + f] = [silu(x), basis0..7] (f16) ----
__global__ void feat1_kernel(const float* __restrict__ X, _Float16* __restrict__ F, int n) {
  int idx = blockIdx.x * blockDim.x + threadIdx.x;
  if (idx >= n) return;
  float x = X[idx];
  float bas[8];
  bspline8(x, bas);
  _Float16* p = F + (long)idx * FEAT;
  p[0] = (_Float16)silu_f(x);
  #pragma unroll
  for (int g = 0; g < 8; ++g) p[1 + g] = (_Float16)bas[g];
}

// ---- layer-1 weights, transposed+prescaled: Wt[(o*IN1+i)*9+f] ----
__global__ void prep_w1(const float* __restrict__ coef, const float* __restrict__ sb,
                        const float* __restrict__ ssp, _Float16* __restrict__ Wt, int n) {
  int idx = blockIdx.x * blockDim.x + threadIdx.x;   // idx = o*IN1 + i
  if (idx >= n) return;
  float s_b = sb[idx];
  float s_p = ssp[idx];
  const float* c = coef + (long)idx * NB;
  _Float16* p = Wt + (long)idx * FEAT;
  p[0] = (_Float16)s_b;
  #pragma unroll
  for (int g = 0; g < 8; ++g) p[1 + g] = (_Float16)(s_p * c[g]);
}

// ---- layer-2 weights, rows 10..15 zero-padded ----
__global__ void prep_w2(const float* __restrict__ coef, const float* __restrict__ sb,
                        const float* __restrict__ ssp, _Float16* __restrict__ Wt) {
  int idx = blockIdx.x * blockDim.x + threadIdx.x;   // idx = o*H1 + i, o in [0,16)
  if (idx >= NOUT2P * H1) return;
  int o = idx / H1;
  _Float16* p = Wt + (long)idx * FEAT;
  if (o < OUT2) {
    float s_b = sb[idx];
    float s_p = ssp[idx];
    const float* c = coef + (long)idx * NB;
    p[0] = (_Float16)s_b;
    #pragma unroll
    for (int g = 0; g < 8; ++g) p[1 + g] = (_Float16)(s_p * c[g]);
  } else {
    #pragma unroll
    for (int f = 0; f < FEAT; ++f) p[f] = (_Float16)0.0f;
  }
}

// ---- SELU + layer-2 features ----
__global__ void feat2_kernel(const float* __restrict__ Hraw, _Float16* __restrict__ F, int n) {
  int idx = blockIdx.x * blockDim.x + threadIdx.x;
  if (idx >= n) return;
  float x = selu_f(Hraw[idx]);
  float bas[8];
  bspline8(x, bas);
  _Float16* p = F + (long)idx * FEAT;
  p[0] = (_Float16)silu_f(x);
  #pragma unroll
  for (int g = 0; g < 8; ++g) p[1 + g] = (_Float16)bas[g];
}

// Load one 16x32 f16 A-subtile; contiguous k order (k = 16h + s), same as B.
// lane<16: row=lane, K[0..15]; lane>=16: row=lane-16, K[16..31]
__device__ __forceinline__ v16h load_a_tile(const _Float16* p) {
  v8h a0 = *(const v8h*)(p);
  v8h a1 = *(const v8h*)(p + 8);
  return __builtin_shufflevector(a0, a1, 0,1,2,3,4,5,6,7,8,9,10,11,12,13,14,15);
}

// ---- 2x2 register-tiled split-K WMMA GEMM for layer 1 ----
// Each wave computes a 32(M) x 32(N) output block over one K-chunk:
// 4 independent v_wmma_f32_16x16x32_f16 per K=32 step, 2x operand reuse.
__global__ void __launch_bounds__(128)
wmma_splitk_gemm_2x2(const _Float16* __restrict__ A, const _Float16* __restrict__ W,
                     float* __restrict__ C, int K, int ntiles2, int tiles, int kchunk, int ldc) {
  const int lane  = threadIdx.x & 31;
  const int gwave = (blockIdx.x * 128 + threadIdx.x) >> 5;
  const int tile  = gwave % tiles;        // 2x2 macro-tile id
  const int split = gwave / tiles;
  const int mt2 = tile / ntiles2;         // 32-row block
  const int nt2 = tile % ntiles2;         // 32-col block
  const int hl = lane & 15;
  const bool hi = lane >= 16;

  const long k0 = (long)split * kchunk;
  const _Float16* arow0 = A + (long)(mt2 * 32 + hl) * K + k0 + (hi ? 16 : 0);
  const _Float16* arow1 = arow0 + (long)16 * K;
  const _Float16* brow0 = W + (long)(nt2 * 32 + hl) * K + k0 + (hi ? 16 : 0);
  const _Float16* brow1 = brow0 + (long)16 * K;

  v8f acc00 = {}, acc01 = {}, acc10 = {}, acc11 = {};
  #pragma unroll 2
  for (int kk = 0; kk < kchunk; kk += 32) {
    v16h a0 = load_a_tile(arow0 + kk);
    v16h a1 = load_a_tile(arow1 + kk);
    v16h b0 = *(const v16h*)(brow0 + kk);   // K[base..base+15] contiguous, 32B aligned
    v16h b1 = *(const v16h*)(brow1 + kk);
    acc00 = __builtin_amdgcn_wmma_f32_16x16x32_f16(false, a0, false, b0, (short)0, acc00, false, false);
    acc01 = __builtin_amdgcn_wmma_f32_16x16x32_f16(false, a0, false, b1, (short)0, acc01, false, false);
    acc10 = __builtin_amdgcn_wmma_f32_16x16x32_f16(false, a1, false, b0, (short)0, acc10, false, false);
    acc11 = __builtin_amdgcn_wmma_f32_16x16x32_f16(false, a1, false, b1, (short)0, acc11, false, false);
  }

  // C/D layout: VGPR r -> row (msub + r + (lane>=16 ? 8:0)), col nsub + lane%16
  const int rbase0 = mt2 * 32 + (hi ? 8 : 0);
  const int cbase0 = nt2 * 32 + hl;
  float* c00 = C + (long)rbase0 * ldc + cbase0;
  #pragma unroll
  for (int r = 0; r < 8; ++r) {
    atomicAdd(c00 + (long)r * ldc,               acc00[r]);
    atomicAdd(c00 + (long)r * ldc + 16,          acc01[r]);
    atomicAdd(c00 + (long)(r + 16) * ldc,        acc10[r]);
    atomicAdd(c00 + (long)(r + 16) * ldc + 16,   acc11[r]);
  }
}

// ---- simple 16x16 split-K WMMA GEMM (layer 2, tiny) ----
__global__ void __launch_bounds__(128)
wmma_splitk_gemm(const _Float16* __restrict__ A, const _Float16* __restrict__ W,
                 float* __restrict__ C, int K, int ntiles, int tiles, int kchunk, int ldc) {
  const int lane  = threadIdx.x & 31;
  const int gwave = (blockIdx.x * 128 + threadIdx.x) >> 5;
  const int tile  = gwave % tiles;
  const int split = gwave / tiles;
  const int mt = tile / ntiles;
  const int nt = tile % ntiles;
  const int hl = lane & 15;
  const bool hi = lane >= 16;

  const long k0 = (long)split * kchunk;
  const _Float16* arow = A + (long)(mt * 16 + hl) * K + k0 + (hi ? 16 : 0);
  const _Float16* brow = W + (long)(nt * 16 + hl) * K + k0 + (hi ? 16 : 0);

  v8f acc = {};
  #pragma unroll 4
  for (int kk = 0; kk < kchunk; kk += 32) {
    v16h av = load_a_tile(arow + kk);
    v16h bv = *(const v16h*)(brow + kk);
    acc = __builtin_amdgcn_wmma_f32_16x16x32_f16(false, av, false, bv, (short)0, acc, false, false);
  }

  float* crow = C + (long)(mt * 16 + (hi ? 8 : 0)) * ldc + nt * 16 + hl;
  #pragma unroll
  for (int r = 0; r < 8; ++r)
    atomicAdd(crow + (long)r * ldc, acc[r]);
}

// ---- final copy: drop the N-padding ----
__global__ void copy_out(const float* __restrict__ O2, float* __restrict__ out) {
  int idx = blockIdx.x * blockDim.x + threadIdx.x;
  if (idx >= B_SZ * OUT2) return;
  int b = idx / OUT2, o = idx % OUT2;
  out[idx] = O2[b * NOUT2P + o];
}

extern "C" void kernel_launch(void* const* d_in, const int* in_sizes, int n_in,
                              void* d_out, int out_size, void* d_ws, size_t ws_size,
                              hipStream_t stream) {
  const float* x     = (const float*)d_in[0];
  const float* coef1 = (const float*)d_in[1];
  const float* sb1   = (const float*)d_in[2];
  const float* ssp1  = (const float*)d_in[3];
  const float* coef2 = (const float*)d_in[4];
  const float* sb2   = (const float*)d_in[5];
  const float* ssp2  = (const float*)d_in[6];
  float* out = (float*)d_out;

  char* ws = (char*)d_ws;
  _Float16* feat1 = (_Float16*)(ws);              // 128*27648 f16 =  7,077,888 B
  _Float16* w1t   = (_Float16*)(ws + 7077888);    // 256*27648 f16 = 14,155,776 B
  float*    h     = (float*)   (ws + 21233664);   // 128*256 f32   =    131,072 B
  _Float16* feat2 = (_Float16*)(ws + 21364736);   // 128*2304 f16  =    589,824 B
  _Float16* w2t   = (_Float16*)(ws + 21954560);   // 16*2304 f16   =     73,728 B
  float*    o2    = (float*)   (ws + 22028288);   // 128*16 f32    =      8,192 B

  hipMemsetAsync(h,  0, (size_t)B_SZ * H1 * sizeof(float), stream);
  hipMemsetAsync(o2, 0, (size_t)B_SZ * NOUT2P * sizeof(float), stream);

  feat1_kernel<<<(B_SZ * IN1) / 256, 256, 0, stream>>>(x, feat1, B_SZ * IN1);
  prep_w1<<<(H1 * IN1) / 256, 256, 0, stream>>>(coef1, sb1, ssp1, w1t, H1 * IN1);
  prep_w2<<<(NOUT2P * H1) / 256, 256, 0, stream>>>(coef2, sb2, ssp2, w2t);

  // GEMM1: M=128, N=256, K=27648 -> (4x8)=32 macro-tiles of 32x32, 27 K-splits
  //        = 864 waves, 4 WMMA per K-step, ~110 MB L2 traffic (2x better reuse)
  wmma_splitk_gemm_2x2<<<216, 128, 0, stream>>>(feat1, w1t, h, K1, 8, 32, 1024, H1);

  feat2_kernel<<<(B_SZ * H1) / 256, 256, 0, stream>>>(h, feat2, B_SZ * H1);

  // GEMM2: M=128, N=16, K=2304 -> 8 tiles x 9 K-splits = 72 waves
  wmma_splitk_gemm<<<18, 128, 0, stream>>>(feat2, w2t, o2, K2, 1, 8, 256, NOUT2P);

  copy_out<<<5, 256, 0, stream>>>(o2, out);
}
